// AitchisonAitkenKernel_53506702573581
// MI455X (gfx1250) — matrix operand order repaired and checked by reading, not compiled, not run on hardware
//
#include <hip/hip_runtime.h>
#include <hip/hip_bf16.h>
#include <math.h>

typedef __attribute__((ext_vector_type(16))) _Float16 v16h;
typedef __attribute__((ext_vector_type(8)))  _Float16 v8h;
typedef __attribute__((ext_vector_type(4)))  _Float16 v4h;
typedef __attribute__((ext_vector_type(8)))  float    v8f;

#define NROWS 8192   // n_test
#define NCOLS 8192   // n_train
#define NF    64     // features
#define NLVL  4
#define KEXP  256    // expanded K = NF * NLVL

// ---------------------------------------------------------------------------
// Kernel 1: w_hi/w_res (f16 split of log(1-h)-log(h/3)), base scalar,
// rowmax[] = -inf.
// ---------------------------------------------------------------------------
__global__ void aa_init_kernel(const float* __restrict__ bw,
                               _Float16* __restrict__ wh,
                               _Float16* __restrict__ wr,
                               float* __restrict__ basep,
                               float* __restrict__ rowmax) {
    int gid = blockIdx.x * blockDim.x + threadIdx.x;
    if (gid < NROWS) rowmax[gid] = -INFINITY;
    if (gid < NF) {
        float h = bw[gid];
        float w = logf(1.0f - h) - logf(h / (float)(NLVL - 1));
        _Float16 hi = (_Float16)w;
        wh[gid] = hi;
        wr[gid] = (_Float16)(w - (float)hi);
    }
    if (gid == 0) {
        float s_false = 0.0f, s_h = 0.0f;
        for (int f = 0; f < NF; ++f) {
            float h = bw[f];
            s_false += logf(h / (float)(NLVL - 1));
            s_h     += logf(h);
        }
        *basep = s_false - s_h;
    }
}

// ---------------------------------------------------------------------------
// Kernel 2a/2b: one-hot expansion into fragment-ready row-major f16:
//   Ah[i, 4f+l] = (test[i,f]==l) ? w_hi[f] : 0
//   Ar[i, 4f+l] = (test[i,f]==l) ? w_res[f] : 0
//   Bx[j, 4f+l] = (train[j,f]==l) ? 1 : 0
// One thread per (row, feature); writes 8-byte v4h chunks.
// ---------------------------------------------------------------------------
__global__ __launch_bounds__(256)
void aa_expand_test(const float* __restrict__ testX,
                    const _Float16* __restrict__ wh,
                    const _Float16* __restrict__ wr,
                    _Float16* __restrict__ Ah,
                    _Float16* __restrict__ Ar) {
    const int gid = blockIdx.x * blockDim.x + threadIdx.x;  // row*64 + f
    const int f = gid & (NF - 1);
    const int tv = (int)testX[gid];
    const _Float16 whf = wh[f], wrf = wr[f];
    v4h hv, rv;
    #pragma unroll
    for (int l = 0; l < NLVL; ++l) {
        const bool hit = (tv == l);
        hv[l] = hit ? whf : (_Float16)0.0f;
        rv[l] = hit ? wrf : (_Float16)0.0f;
    }
    *reinterpret_cast<v4h*>(Ah + (size_t)gid * 4) = hv;
    *reinterpret_cast<v4h*>(Ar + (size_t)gid * 4) = rv;
}

__global__ __launch_bounds__(256)
void aa_expand_train(const float* __restrict__ trainX,
                     _Float16* __restrict__ Bx) {
    const int gid = blockIdx.x * blockDim.x + threadIdx.x;  // col*64 + f
    const int sv = (int)trainX[gid];
    v4h bv;
    #pragma unroll
    for (int l = 0; l < NLVL; ++l)
        bv[l] = (sv == l) ? (_Float16)1.0f : (_Float16)0.0f;
    *reinterpret_cast<v4h*>(Bx + (size_t)gid * 4) = bv;
}

// ---------------------------------------------------------------------------
__device__ __forceinline__ void atomicMaxF(float* addr, float val) {
    if (val >= 0.0f)
        atomicMax((int*)addr, __float_as_int(val));
    else
        atomicMin((unsigned int*)addr, __float_as_uint(val));
}

// ---------------------------------------------------------------------------
// Kernel 3: WMMA GEMM, match = A @ B^T over expanded K=256, f32 accum.
// Block = 256 threads (8 waves) -> tile 128M x 128N. B tile (64 KB, a
// contiguous slab of Bx) staged into LDS, reused by all 8 waves.
// Each wave: 16M x 128N strip = 8 accumulators; per k-step:
//   4 global_load_b128 (A hi/res) + 16 ds_load_b128 (B) + 16 v_wmma.
// A-fragment lane chunks: expanded K = 32ks + 8g + {0..7} and +16.
// B-fragment lane chunks: expanded K = 32ks + 8g + {0..7} and +16.
// ---------------------------------------------------------------------------
__global__ __launch_bounds__(256)
void aa_gemm_kernel(const _Float16* __restrict__ Ah,
                    const _Float16* __restrict__ Ar,
                    const _Float16* __restrict__ Bx,
                    const float* __restrict__ basep,
                    float* __restrict__ out,
                    float* __restrict__ rowmax) {
    __shared__ _Float16 sB[128 * KEXP];   // 64 KB

    const int tid  = threadIdx.x;
    const int lane = tid & 31;
    const int wave = tid >> 5;
    const int g    = lane >> 4;
    const int m    = lane & 15;

    const int col0 = blockIdx.x * 128;
    const int row0 = blockIdx.y * 128;

    // ---- cooperative stage: 128 consecutive Bx rows = contiguous 64 KB
    {
        const uint4* src = reinterpret_cast<const uint4*>(Bx + (size_t)col0 * KEXP);
        uint4* dst = reinterpret_cast<uint4*>(sB);
        #pragma unroll
        for (int i = 0; i < 16; ++i)
            dst[tid + i * 256] = src[tid + i * 256];
    }
    __syncthreads();

    const int rowA = row0 + wave * 16 + m;
    const _Float16* __restrict__ arow_h = Ah + (size_t)rowA * KEXP;
    const _Float16* __restrict__ arow_r = Ar + (size_t)rowA * KEXP;

    v8f acc[8] = {};

    #pragma unroll
    for (int ks = 0; ks < 8; ++ks) {
        const int k0 = 32 * ks + 8 * g;          // A chunk base (halves)
        const v8h h0 = *reinterpret_cast<const v8h*>(arow_h + k0);
        const v8h h1 = *reinterpret_cast<const v8h*>(arow_h + k0 + 16);
        const v8h r0 = *reinterpret_cast<const v8h*>(arow_r + k0);
        const v8h r1 = *reinterpret_cast<const v8h*>(arow_r + k0 + 16);
        const v16h ah = __builtin_shufflevector(h0, h1,
            0, 1, 2, 3, 4, 5, 6, 7, 8, 9, 10, 11, 12, 13, 14, 15);
        const v16h ar = __builtin_shufflevector(r0, r1,
            0, 1, 2, 3, 4, 5, 6, 7, 8, 9, 10, 11, 12, 13, 14, 15);

        const int kb = 32 * ks + 8 * g;          // B chunk base (halves)
        #pragma unroll
        for (int half = 0; half < 2; ++half) {
            v16h b[4];
            #pragma unroll
            for (int t = 0; t < 4; ++t) {
                const int col = 16 * (4 * half + t) + m;
                const v8h b0 = *reinterpret_cast<const v8h*>(sB + col * KEXP + kb);
                const v8h b1 = *reinterpret_cast<const v8h*>(sB + col * KEXP + kb + 16);
                b[t] = __builtin_shufflevector(b0, b1,
                    0, 1, 2, 3, 4, 5, 6, 7, 8, 9, 10, 11, 12, 13, 14, 15);
            }
            #pragma unroll
            for (int t = 0; t < 4; ++t)
                acc[4 * half + t] = __builtin_amdgcn_wmma_f32_16x16x32_f16(
                    false, ah, false, b[t], (short)0, acc[4 * half + t], false, false);
            #pragma unroll
            for (int t = 0; t < 4; ++t)
                acc[4 * half + t] = __builtin_amdgcn_wmma_f32_16x16x32_f16(
                    false, ar, false, b[t], (short)0, acc[4 * half + t], false, false);
        }
    }

    // ---- Epilogue: write log_dist, fold per-row maxima.
    // D layout: VGPR v of lane (g,n) holds (M = v + 8*g, N = n).
    const float base = *basep;
    #pragma unroll
    for (int v = 0; v < 8; ++v) {
        const int orow = row0 + wave * 16 + v + 8 * g;
        float mx = -INFINITY;
        #pragma unroll
        for (int t = 0; t < 8; ++t) {
            const float ld = acc[t][v] + base;
            out[(size_t)orow * NCOLS + (col0 + 16 * t + m)] = ld;
            mx = fmaxf(mx, ld);
        }
        #pragma unroll
        for (int s = 1; s < 16; s <<= 1)
            mx = fmaxf(mx, __shfl_xor(mx, s, 32));
        if (m == 0)
            atomicMaxF(rowmax + orow, mx);
    }
}

// ---------------------------------------------------------------------------
// Kernel 4: out[i,j] = rowmax[i] * exp(log_dist[i,j] - rowmax[i]) in place.
// ---------------------------------------------------------------------------
__global__ __launch_bounds__(256)
void aa_finalize_kernel(float* __restrict__ out,
                        const float* __restrict__ rowmax) {
    const int gid = blockIdx.x * blockDim.x + threadIdx.x;   // 16M float4s
    const int r = gid >> 11;                                 // 2048 float4/row
    float4* p = reinterpret_cast<float4*>(out) + gid;
    float4 v = *p;
    const float mlog = rowmax[r];
    v.x = mlog * __expf(v.x - mlog);
    v.y = mlog * __expf(v.y - mlog);
    v.z = mlog * __expf(v.z - mlog);
    v.w = mlog * __expf(v.w - mlog);
    *p = v;
}

// ---------------------------------------------------------------------------
extern "C" void kernel_launch(void* const* d_in, const int* in_sizes, int n_in,
                              void* d_out, int out_size, void* d_ws, size_t ws_size,
                              hipStream_t stream) {
    (void)in_sizes; (void)n_in; (void)out_size; (void)ws_size;

    const float* bw     = (const float*)d_in[0];  // bandwidths [64]
    const float* testX  = (const float*)d_in[1];  // [8192, 64]
    const float* trainX = (const float*)d_in[2];  // [8192, 64]
    float* out = (float*)d_out;                   // [8192, 8192]

    // workspace: wh|wr|base|rowmax (~33 KB), then 3x 4MB expanded operands
    char* ws = (char*)d_ws;
    _Float16* wh     = (_Float16*)(ws + 0);
    _Float16* wr     = (_Float16*)(ws + 128);
    float*    basep  = (float*)   (ws + 256);
    float*    rowmax = (float*)   (ws + 512);
    const size_t MB4 = (size_t)NROWS * KEXP * sizeof(_Float16);  // 4 MB
    _Float16* Ah = (_Float16*)(ws + 65536);
    _Float16* Ar = (_Float16*)(ws + 65536 + MB4);
    _Float16* Bx = (_Float16*)(ws + 65536 + 2 * MB4);

    aa_init_kernel<<<NROWS / 256, 256, 0, stream>>>(bw, wh, wr, basep, rowmax);

    const int nexp = NROWS * NF;  // 512K threads
    aa_expand_test <<<nexp / 256, 256, 0, stream>>>(testX, wh, wr, Ah, Ar);
    aa_expand_train<<<nexp / 256, 256, 0, stream>>>(trainX, Bx);

    dim3 grid(NCOLS / 128, NROWS / 128);
    aa_gemm_kernel<<<grid, 256, 0, stream>>>(Ah, Ar, Bx, basep, out, rowmax);

    const int n4 = (NROWS * (NCOLS / 4));
    aa_finalize_kernel<<<n4 / 256, 256, 0, stream>>>(out, rowmax);
}
